// GPNN_Event_592705487034
// MI455X (gfx1250) — compile-verified, matrix operand order checked by reading, not run on hardware
//
#include <hip/hip_runtime.h>

#define BB 32
#define NN 128
#define PP 8128
#define R3 3

typedef __attribute__((ext_vector_type(16))) __bf16 v16bf;
typedef __attribute__((ext_vector_type(8)))  float  v8f;
typedef __attribute__((ext_vector_type(8)))  unsigned short u16x8;
typedef int v4i __attribute__((__vector_size__(4 * sizeof(int))));

// ---- async global<->LDS copies (gfx1250), with synchronous fallback ----
#if defined(__has_builtin)
#if __has_builtin(__builtin_amdgcn_global_load_async_to_lds_b128)
#define HAVE_ASYNC_LD 1
#else
#define HAVE_ASYNC_LD 0
#endif
#if __has_builtin(__builtin_amdgcn_global_store_async_from_lds_b128)
#define HAVE_ASYNC_ST 1
#else
#define HAVE_ASYNC_ST 0
#endif
#if __has_builtin(__builtin_amdgcn_s_wait_asynccnt)
#define HAVE_WAIT_ASYNC 1
#else
#define HAVE_WAIT_ASYNC 0
#endif
#else
#define HAVE_ASYNC_LD 0
#define HAVE_ASYNC_ST 0
#define HAVE_WAIT_ASYNC 0
#endif

__device__ __forceinline__ void cp16_g2l(void* lds, const void* g){
#if HAVE_ASYNC_LD
  __builtin_amdgcn_global_load_async_to_lds_b128(
      (__attribute__((address_space(1))) v4i*)(size_t)g,
      (__attribute__((address_space(3))) v4i*)(unsigned)(size_t)lds, 0, 0);
#else
  *(u16x8*)lds = *(const u16x8*)g;
#endif
}
__device__ __forceinline__ void cp16_l2g(void* g, const void* lds){
#if HAVE_ASYNC_ST
  __builtin_amdgcn_global_store_async_from_lds_b128(
      (__attribute__((address_space(1))) v4i*)(size_t)g,
      (__attribute__((address_space(3))) v4i*)(unsigned)(size_t)lds, 0, 0);
#else
  *(u16x8*)g = *(const u16x8*)lds;
#endif
}
__device__ __forceinline__ void wait_async0(){
#if HAVE_ASYNC_LD || HAVE_ASYNC_ST
#if HAVE_WAIT_ASYNC
  __builtin_amdgcn_s_wait_asynccnt(0);
#else
  asm volatile("s_wait_asynccnt 0x0" ::: "memory");
#endif
#endif
}

__device__ __forceinline__ unsigned short f2bf(float f){
  unsigned u = __float_as_uint(f);
  u += 0x7FFFu + ((u >> 16) & 1u);
  return (unsigned short)(u >> 16);
}
__device__ __forceinline__ float bf2f(unsigned short h){
  return __uint_as_float(((unsigned)h) << 16);
}
__device__ __forceinline__ v8f vzero8(){
  v8f v;
#pragma unroll
  for (int i = 0; i < 8; ++i) v[i] = 0.f;
  return v;
}

// 16-bit A-operand fragment per CDNA5 ISA layout:
// lanes 0-15: M=lane, K = k0+0..7 then k0+16..23
// lanes 16-31: M=lane-16, K = k0+8..15 then k0+24..31
// B fragment = A fragment of B^T (weights stored transposed, [col][k]).
__device__ __forceinline__ v16bf load_frag(const unsigned short* p0, int row, int k0, int stride){
  int lane = threadIdx.x & 31;
  const unsigned short* p = p0 + (size_t)(row + (lane & 15)) * stride
                               + (k0 + ((lane & 16) ? 8 : 0));
  union { v16bf v; u16x8 h[2]; } u;
  u.h[0] = *(const u16x8*)p;
  u.h[1] = *(const u16x8*)(p + 16);
  return u.v;
}

__device__ __forceinline__ v8f wmma_bf16(v16bf a, v16bf b, v8f c){
  return __builtin_amdgcn_wmma_f32_16x16x32_bf16(false, a, false, b, (short)0, c, false, false);
}

// ---------------- weight prep: fp32 (K,C) -> bf16 transposed [c*K + k] ----------------
__global__ void transpose_bf16(const float* __restrict__ src, unsigned short* __restrict__ dst,
                               int K, int C){
  int i = blockIdx.x * 256 + threadIdx.x;
  if (i >= K * C) return;
  int c = i / K, k = i - c * K;
  dst[i] = f2bf(src[(size_t)k * C + c]);
}

// ---------------- e = emb[edge_ids] as bf16 ----------------
__global__ void init_e_kernel(const int* __restrict__ ids, const float* __restrict__ emb,
                              unsigned short* __restrict__ e){
  size_t idx = (size_t)blockIdx.x * 256 + threadIdx.x;   // one thread per 8 elems
  size_t edge = idx >> 4;
  int c8 = (int)(idx & 15) << 3;
  int id = ids[edge];
#pragma unroll
  for (int q = 0; q < 8; ++q)
    e[edge * 128 + c8 + q] = f2bf(emb[id * 128 + c8 + q]);
}

// ---------------- generic (4096 x CC) = A(4096x128) @ W(128xCC), WT bf16 [c][k] ----------------
template <int CC>
__global__ __launch_bounds__(256) void gemm_k128(const float* __restrict__ A,
                                                 const unsigned short* __restrict__ WT,
                                                 float* __restrict__ Out){
  constexpr int CW = CC / 8;     // cols per wave (64 or 48)
  constexpr int NT = CW / 16;    // col tiles per wave (4 or 3)
  __shared__ __align__(16) unsigned short sA[16 * 136];
  const int tid = threadIdx.x;
  const int r0g = blockIdx.x << 4;
  for (int i = tid; i < 16 * 128; i += 256){
    int r = i >> 7, c = i & 127;
    sA[r * 136 + c] = f2bf(A[(size_t)(r0g + r) * 128 + c]);
  }
  __syncthreads();
  const int lane = tid & 31, wv = tid >> 5;
  const int c0 = wv * CW;
  v16bf aF[4];
#pragma unroll
  for (int k = 0; k < 4; ++k) aF[k] = load_frag(sA, 0, 32 * k, 136);
  v8f acc[NT];
#pragma unroll
  for (int t = 0; t < NT; ++t) acc[t] = vzero8();
#pragma unroll
  for (int k = 0; k < 4; ++k){
#pragma unroll
    for (int t = 0; t < NT; ++t){
      v16bf bf = load_frag(WT, c0 + 16 * t, 32 * k, 128);
      acc[t] = wmma_bf16(aF[k], bf, acc[t]);
    }
  }
  const int colb = lane & 15;
  const int hi8 = (lane & 16) ? 8 : 0;
#pragma unroll
  for (int t = 0; t < NT; ++t){
#pragma unroll
    for (int j = 0; j < 8; ++j)
      Out[(size_t)(r0g + j + hi8) * CC + c0 + 16 * t + colb] = acc[t][j];
  }
}

// ---------------- fused per-edge round kernel: one workgroup per (b, w) ----------------
__global__ __launch_bounds__(256) void edge_kernel(
    unsigned short* __restrict__ eG,
    const unsigned short* __restrict__ WleT,
    const unsigned short* __restrict__ WmeT,
    const unsigned short* __restrict__ WueT,
    const unsigned short* __restrict__ WumT,
    const float* __restrict__ hprec,   // (B*N, 512): [hlw | hlv | hmw | hmv]
    const int*  __restrict__ evn,
    const float* __restrict__ bl1,
    const float* __restrict__ wl2,
    const float* __restrict__ bl2,
    const float* __restrict__ bm,
    const float* __restrict__ bu,
    float* __restrict__ ms)
{
  __shared__ __align__(16) unsigned short sA[128 * 136];  // e tile -> m tile -> e_new tile
  __shared__ __align__(16) unsigned short sW[64 * 136];   // weight half (64 cols x 128 k)
  const int b = blockIdx.x >> 7;
  const int w = blockIdx.x & 127;
  const int evb = evn[b];
  if (w >= evb) return;        // whole (b,w) row invalid: e unchanged, m == 0

  const int tid = threadIdx.x;
  unsigned short* eRow = eG + ((size_t)(b * NN + w)) * NN * 128;
  for (int i = tid; i < 128 * 16; i += 256){
    int r = i >> 4, c8 = (i & 15) << 3;
    cp16_g2l(&sA[r * 136 + c8], eRow + r * 128 + c8);
  }
  wait_async0();
  __syncthreads();

  const int lane = tid & 31, wv = tid >> 5;
  const int r0 = wv << 4;                 // this wave's 16 v-rows
  const int colb = lane & 15;
  const int hi8 = (lane & 16) ? 8 : 0;

  v16bf aF[4];
#pragma unroll
  for (int k = 0; k < 4; ++k) aF[k] = load_frag(sA, r0, 32 * k, 136);

  v8f acc[8];
  // fully unrolled so every acc index is a literal (no v_movrel indexing)
  auto run_gemm = [&](const unsigned short* __restrict__ WT, const v16bf* af, bool zero){
    if (zero){
#pragma unroll
      for (int t = 0; t < 8; ++t) acc[t] = vzero8();
    }
#pragma unroll
    for (int half = 0; half < 2; ++half){
      __syncthreads();
      for (int i = tid; i < 64 * 16; i += 256){
        int r = i >> 4, c8 = (i & 15) << 3;
        cp16_g2l(&sW[r * 136 + c8], WT + (size_t)(half * 64 + r) * 128 + c8);
      }
      wait_async0();
      __syncthreads();
#pragma unroll
      for (int k = 0; k < 4; ++k){
#pragma unroll
        for (int t2 = 0; t2 < 4; ++t2){
          v16bf bf = load_frag(sW, 16 * t2, 32 * k, 136);
          acc[half * 4 + t2] = wmma_bf16(af[k], bf, acc[half * 4 + t2]);
        }
      }
    }
  };

  const size_t hb = (size_t)(b * NN) * 512;

  // ---- stage 1: z = relu(e@Wl_e + hlw[w] + hlv[v] + bl1); adj = z.wl2 + bl2; s = sigmoid
  run_gemm(WleT, aF, true);
  float cw[8], wl2v[8];
#pragma unroll
  for (int t = 0; t < 8; ++t){
    int c = 16 * t + colb;
    cw[t] = hprec[hb + (size_t)w * 512 + c] + bl1[c];
    wl2v[t] = wl2[c];
  }
  float part[8];
#pragma unroll
  for (int j = 0; j < 8; ++j) part[j] = 0.f;
#pragma unroll
  for (int t = 0; t < 8; ++t){
#pragma unroll
    for (int j = 0; j < 8; ++j){
      int row = r0 + j + hi8;
      float z = acc[t][j] + cw[t] + hprec[hb + (size_t)row * 512 + 128 + 16 * t + colb];
      part[j] = fmaf(fmaxf(z, 0.f), wl2v[t], part[j]);
    }
  }
#pragma unroll
  for (int msk = 1; msk < 16; msk <<= 1){
#pragma unroll
    for (int j = 0; j < 8; ++j) part[j] += __shfl_xor(part[j], msk, 32);
  }
  const float bl2v = bl2[0];
  float sv[8];
#pragma unroll
  for (int j = 0; j < 8; ++j) sv[j] = 1.f / (1.f + __expf(-(part[j] + bl2v)));

  // ---- stage 2: m = relu(e@Wm_e + hmw[w] + hmv[v] + bm) * s, masked; write to LDS
  run_gemm(WmeT, aF, true);
#pragma unroll
  for (int t = 0; t < 8; ++t){
    int c = 16 * t + colb;
    cw[t] = hprec[hb + (size_t)w * 512 + 256 + c] + bm[c];
  }
  __syncthreads();
#pragma unroll
  for (int t = 0; t < 8; ++t){
#pragma unroll
    for (int j = 0; j < 8; ++j){
      int row = r0 + j + hi8;
      float mv = fmaxf(acc[t][j] + cw[t] + hprec[hb + (size_t)row * 512 + 384 + 16 * t + colb], 0.f) * sv[j];
      if (row >= evb) mv = 0.f;
      sA[row * 136 + 16 * t + colb] = f2bf(mv);
    }
  }
  __syncthreads();
  // ms[b, v, :] += m[b, w, v, :]  (cross-w reduction via L2 atomics)
  float* msRow = ms + (size_t)(b * NN) * 128;
  for (int i = tid; i < 128 * 128; i += 256){
    int v = i >> 7, c = i & 127;
    float mv = bf2f(sA[v * 136 + c]);
    if (mv != 0.f) atomicAdd(&msRow[(size_t)v * 128 + c], mv);
  }

  // ---- stage 3+4: e_new = relu(e@Wu_e + m@Wu_m + bu) for valid rows
  run_gemm(WueT, aF, true);
  v16bf mF[4];
#pragma unroll
  for (int k = 0; k < 4; ++k) mF[k] = load_frag(sA, r0, 32 * k, 136);
  run_gemm(WumT, mF, false);
  __syncthreads();
#pragma unroll
  for (int t = 0; t < 8; ++t){
#pragma unroll
    for (int j = 0; j < 8; ++j){
      int row = r0 + j + hi8;
      if (row < evb){
        float ev_ = fmaxf(acc[t][j] + bu[16 * t + colb], 0.f);
        sA[row * 136 + 16 * t + colb] = f2bf(ev_);
      }
    }
  }
  __syncthreads();
  for (int i = tid; i < 128 * 16; i += 256){
    int r = i >> 4, c8 = (i & 15) << 3;
    if (r < evb)
      cp16_l2g(eRow + r * 128 + c8, &sA[r * 136 + c8]);
  }
  // ENDPGM implicitly waits all counters; async stores complete before kernel end.
}

// ---------------- GRU elementwise ----------------
__global__ void gru_kernel(const float* __restrict__ gi, const float* __restrict__ gh,
                           const float* __restrict__ b_ih, const float* __restrict__ b_hh,
                           const int* __restrict__ evn, float* __restrict__ h){
  int bn = blockIdx.x;
  int b = bn >> 7, n = bn & 127;
  int c = threadIdx.x;
  if (n >= evn[b]) return;
  size_t i3 = (size_t)bn * 384, i1 = (size_t)bn * 128;
  float ir = gi[i3 + c]       + b_ih[c];
  float iz = gi[i3 + 128 + c] + b_ih[128 + c];
  float in_= gi[i3 + 256 + c] + b_ih[256 + c];
  float hr = gh[i3 + c]       + b_hh[c];
  float hz = gh[i3 + 128 + c] + b_hh[128 + c];
  float hn = gh[i3 + 256 + c] + b_hh[256 + c];
  float r  = 1.f / (1.f + __expf(-(ir + hr)));
  float zg = 1.f / (1.f + __expf(-(iz + hz)));
  float nn = tanhf(in_ + r * hn);
  h[i1 + c] = (1.f - zg) * nn + zg * h[i1 + c];
}

// ---------------- readout: 64 pairs per workgroup ----------------
__device__ __forceinline__ int tri_off(int i){ return i * 127 - ((i * (i - 1)) >> 1); }

__global__ __launch_bounds__(256) void readout_kernel(
    const unsigned short* __restrict__ eG,
    const unsigned short* __restrict__ Wr1T,   // [256 cols][256 k] bf16
    const float* __restrict__ br1,
    const float* __restrict__ Wr2,             // (256,10) f32
    const float* __restrict__ br2,
    const int* __restrict__ evn,
    float* __restrict__ out)                   // (B,5,P,2) zero-initialized
{
  __shared__ __align__(16) unsigned short sF[64 * 264];  // feat -> hidden (bf16)
  __shared__ float sW2[2560];
  __shared__ int sIU[64], sJU[64];
  const int tid = threadIdx.x;
  const int row0 = blockIdx.x << 6;   // P = 8128 is divisible by 64 -> single b per block
  const int b = row0 / PP;
  const int p0 = row0 - b * PP;
  if (tid < 64){
    int p = p0 + tid;
    int lo = 0, hiB = 126;
    while (lo < hiB){ int mid = (lo + hiB + 1) >> 1; if (tri_off(mid) <= p) lo = mid; else hiB = mid - 1; }
    sIU[tid] = lo;
    sJU[tid] = p - tri_off(lo) + lo + 1;
  }
  for (int i = tid; i < 2560; i += 256) sW2[i] = Wr2[i];
  __syncthreads();
  // feat = [e[b,iu,ju,:], e[b,ju,iu,:]]
  for (int i = tid; i < 64 * 32; i += 256){
    int r = i >> 5, c8 = (i & 31) << 3;
    int iu = sIU[r], ju = sJU[r];
    const unsigned short* src = (c8 < 128)
      ? (eG + (((size_t)(b * NN + iu)) * NN + ju) * 128 + c8)
      : (eG + (((size_t)(b * NN + ju)) * NN + iu) * 128 + (c8 - 128));
    cp16_g2l(&sF[r * 264 + c8], src);
  }
  wait_async0();
  __syncthreads();
  const int lane = tid & 31, wv = tid >> 5;
  const int rt = wv >> 1, ch = wv & 1;
  const int rbase = rt << 4, c0 = ch << 7;
  v8f acc[8];
#pragma unroll
  for (int t = 0; t < 8; ++t) acc[t] = vzero8();
#pragma unroll
  for (int k = 0; k < 8; ++k){
    v16bf a = load_frag(sF, rbase, 32 * k, 264);
#pragma unroll
    for (int t = 0; t < 8; ++t){
      v16bf bf = load_frag(Wr1T, c0 + 16 * t, 32 * k, 256);
      acc[t] = wmma_bf16(a, bf, acc[t]);
    }
  }
  const int colb = lane & 15, hi8 = (lane & 16) ? 8 : 0;
  __syncthreads();
#pragma unroll
  for (int t = 0; t < 8; ++t){
#pragma unroll
    for (int j = 0; j < 8; ++j){
      int r = rbase + j + hi8, c = c0 + 16 * t + colb;
      sF[r * 264 + c] = f2bf(fmaxf(acc[t][j] + br1[c], 0.f));
    }
  }
  __syncthreads();
  const int vn = evn[b];
  for (int idx = tid; idx < 640; idx += 256){
    int r = idx / 10, o = idx - r * 10;
    int ju = sJU[r];
    if (ju >= vn) continue;
    int iu = sIU[r];
    float a2 = br2[o];
    for (int hh = 0; hh < 256; ++hh)
      a2 = fmaf(bf2f(sF[r * 264 + hh]), sW2[hh * 10 + o], a2);
    int oidx = iu * vn - ((iu * (iu + 1)) >> 1) + (ju - iu - 1);
    out[(((size_t)b * 5 + (o >> 1)) * PP + oidx) * 2 + (o & 1)] = a2;
  }
}

extern "C" void kernel_launch(void* const* d_in, const int* in_sizes, int n_in,
                              void* d_out, int out_size, void* d_ws, size_t ws_size,
                              hipStream_t stream){
  (void)in_sizes; (void)n_in; (void)ws_size;
  const int*   edge_ids = (const int*)  d_in[0];
  const float* node_f   = (const float*)d_in[1];
  const int*   evn      = (const int*)  d_in[3];
  const float* emb      = (const float*)d_in[4];
  const float* Wl_e = (const float*)d_in[5];
  const float* Wl_w = (const float*)d_in[6];
  const float* Wl_v = (const float*)d_in[7];
  const float* bl1  = (const float*)d_in[8];
  const float* wl2  = (const float*)d_in[9];
  const float* bl2  = (const float*)d_in[10];
  const float* Wm_w = (const float*)d_in[11];
  const float* Wm_v = (const float*)d_in[12];
  const float* Wm_e = (const float*)d_in[13];
  const float* bm   = (const float*)d_in[14];
  const float* Wu_e = (const float*)d_in[15];
  const float* Wu_m = (const float*)d_in[16];
  const float* bu   = (const float*)d_in[17];
  const float* W_ih = (const float*)d_in[18];
  const float* W_hh = (const float*)d_in[19];
  const float* b_ih = (const float*)d_in[20];
  const float* b_hh = (const float*)d_in[21];
  const float* Wr1  = (const float*)d_in[22];
  const float* br1  = (const float*)d_in[23];
  const float* Wr2  = (const float*)d_in[24];
  const float* br2  = (const float*)d_in[25];
  float* out = (float*)d_out;

  char* ws = (char*)d_ws;
  size_t off = 0;
  auto alloc = [&](size_t bytes)->char*{
    char* p = ws + off; off += (bytes + 255) & ~(size_t)255; return p;
  };
  unsigned short* eB    = (unsigned short*)alloc((size_t)BB * NN * NN * 128 * 2); // 134 MB
  float*          msB   = (float*)alloc((size_t)BB * NN * 128 * 4);
  float*          hB    = (float*)alloc((size_t)BB * NN * 128 * 4);
  float*          hprec = (float*)alloc((size_t)BB * NN * 512 * 4);
  float*          giB   = (float*)alloc((size_t)BB * NN * 384 * 4);
  float*          ghB   = (float*)alloc((size_t)BB * NN * 384 * 4);
  unsigned short* WcatT = (unsigned short*)alloc((size_t)4 * 128 * 128 * 2);
  unsigned short* WleT  = (unsigned short*)alloc((size_t)128 * 128 * 2);
  unsigned short* WmeT  = (unsigned short*)alloc((size_t)128 * 128 * 2);
  unsigned short* WueT  = (unsigned short*)alloc((size_t)128 * 128 * 2);
  unsigned short* WumT  = (unsigned short*)alloc((size_t)128 * 128 * 2);
  unsigned short* WihT  = (unsigned short*)alloc((size_t)384 * 128 * 2);
  unsigned short* WhhT  = (unsigned short*)alloc((size_t)384 * 128 * 2);
  unsigned short* Wr1T  = (unsigned short*)alloc((size_t)256 * 256 * 2);

  (void)hipMemsetAsync(d_out, 0, (size_t)out_size * sizeof(float), stream);

  // weight prep (bf16, transposed for B-fragment loads)
  transpose_bf16<<<64, 256, 0, stream>>>(Wl_w, WcatT + 0 * 16384, 128, 128);
  transpose_bf16<<<64, 256, 0, stream>>>(Wl_v, WcatT + 1 * 16384, 128, 128);
  transpose_bf16<<<64, 256, 0, stream>>>(Wm_w, WcatT + 2 * 16384, 128, 128);
  transpose_bf16<<<64, 256, 0, stream>>>(Wm_v, WcatT + 3 * 16384, 128, 128);
  transpose_bf16<<<64, 256, 0, stream>>>(Wl_e, WleT, 128, 128);
  transpose_bf16<<<64, 256, 0, stream>>>(Wm_e, WmeT, 128, 128);
  transpose_bf16<<<64, 256, 0, stream>>>(Wu_e, WueT, 128, 128);
  transpose_bf16<<<64, 256, 0, stream>>>(Wu_m, WumT, 128, 128);
  transpose_bf16<<<192, 256, 0, stream>>>(W_ih, WihT, 128, 384);
  transpose_bf16<<<192, 256, 0, stream>>>(W_hh, WhhT, 128, 384);
  transpose_bf16<<<256, 256, 0, stream>>>(Wr1, Wr1T, 256, 256);

  // e = emb[edge_ids], h = node_features
  init_e_kernel<<<32768, 256, 0, stream>>>(edge_ids, emb, eB);
  (void)hipMemcpyAsync(hB, node_f, (size_t)BB * NN * 128 * 4, hipMemcpyDeviceToDevice, stream);

  for (int r = 0; r < R3; ++r){
    // hprec = h @ [Wl_w | Wl_v | Wm_w | Wm_v]
    gemm_k128<512><<<256, 256, 0, stream>>>(hB, WcatT, hprec);
    (void)hipMemsetAsync(msB, 0, (size_t)BB * NN * 128 * 4, stream);
    edge_kernel<<<BB * NN, 256, 0, stream>>>(eB, WleT, WmeT, WueT, WumT, hprec, evn,
                                             bl1, wl2, bl2, bm, bu, msB);
    gemm_k128<384><<<256, 256, 0, stream>>>(msB, WihT, giB);
    gemm_k128<384><<<256, 256, 0, stream>>>(hB,  WhhT, ghB);
    gru_kernel<<<BB * NN, 128, 0, stream>>>(giB, ghB, b_ih, b_hh, evn, hB);
  }

  readout_kernel<<<(BB * PP) / 64, 256, 0, stream>>>(eB, Wr1T, br1, Wr2, br2, evn, out);
}
